// MMRelativeMultiHeadAttentionModule_47631187313309
// MI455X (gfx1250) — matrix operand order repaired and checked by reading, not compiled, not run on hardware
//
#include <hip/hip_runtime.h>

// ---------------------------------------------------------------------------
// MI455X (gfx1250, wave32) Transformer-XL relative multi-head attention.
// All GEMMs are M=4096,N=512,K=512 -> one WMMA GEMM kernel reused 7x (+2 out).
// Matrix math: v_wmma_f32_16x16x32_bf16 (f32 accumulate).
// GEMM A-tile staging: async global->LDS DMA (ASYNCcnt) when available.
// ---------------------------------------------------------------------------

typedef __attribute__((ext_vector_type(16))) __bf16          v16bf;
typedef __attribute__((ext_vector_type(16))) unsigned short  v16us;
typedef __attribute__((ext_vector_type(8)))  float           v8f;
typedef int v4i __attribute__((vector_size(16)));
typedef __attribute__((address_space(1))) v4i gv4i;   // global v4i
typedef __attribute__((address_space(3))) v4i lv4i;   // LDS v4i
typedef unsigned short u16;
typedef unsigned int   u32;

#if defined(__has_builtin)
#if __has_builtin(__builtin_amdgcn_global_load_async_to_lds_b128)
#define MMREL_ASYNC_LDS 1
#endif
#if __has_builtin(__builtin_amdgcn_s_wait_asynccnt)
#define MMREL_WAIT_BUILTIN 1
#endif
#endif

__device__ __forceinline__ void wait_async0() {
#if defined(MMREL_WAIT_BUILTIN)
  __builtin_amdgcn_s_wait_asynccnt(0);
#elif defined(MMREL_ASYNC_LDS)
  asm volatile("s_wait_asynccnt 0x0" ::: "memory");
#endif
}

constexpr int Bc = 4, Sc = 1024, Dc = 512, Hc = 8, DHc = 64;
constexpr int GM = Bc * Sc;      // 4096
constexpr int GN = Dc;           // 512
constexpr int GK = Dc;           // 512
constexpr int SCS = 1032;        // f32 LDS row stride (bank de-phasing)
constexpr float SCALE = 0.04419417382415922f;   // 1/sqrt(512)

__device__ __forceinline__ u16 bf16_of(float f) {
  u32 u = __builtin_bit_cast(u32, f);
  u32 r = u + 0x7FFFu + ((u >> 16) & 1u);      // round-to-nearest-even
  return (u16)(r >> 16);
}

__device__ __forceinline__ v8f wmma_bf16(v16bf a, v16bf b, v8f c) {
  // (neg_a, A, neg_b, B, c_mod, C, reuse_a, reuse_b)
  return __builtin_amdgcn_wmma_f32_16x16x32_bf16(false, a, false, b,
                                                 (short)0, c, false, false);
}

// A fragment (16x32 bf16, M x K) from bf16 LDS. Layout: lane = {m=lane&15,
// half=lane>>4}; elements 0..15 -> k = 16*half + e (matches B fragments).
// 16 contiguous u16 per lane.
__device__ __forceinline__ v16bf a_frag_lds(const u16* p0, int stride, int lane) {
  const u16* p = p0 + (lane & 15) * stride + 16 * (lane >> 4);
  v16us r;
#pragma unroll
  for (int e = 0; e < 8; ++e) r[e] = p[e];
#pragma unroll
  for (int e = 0; e < 8; ++e) r[8 + e] = p[8 + e];
  return __builtin_bit_cast(v16bf, r);
}

// Same A-fragment geometry from an f32 LDS tile (async-DMA'd), converting
// to bf16 at fragment-load time. Two contiguous 32B chunks -> ds_load_b128 x4.
__device__ __forceinline__ v16bf a_frag_lds_f32(const float* p0, int stride, int lane) {
  const float* p = p0 + (lane & 15) * stride + 16 * (lane >> 4);
  v16us r;
#pragma unroll
  for (int e = 0; e < 8; ++e) r[e] = bf16_of(p[e]);
#pragma unroll
  for (int e = 0; e < 8; ++e) r[8 + e] = bf16_of(p[8 + e]);
  return __builtin_bit_cast(v16bf, r);
}

// B fragment (32x16 bf16, K x N): lane n = lane&15, k = e + 16*(lane>>4):
// 16 contiguous u16 starting at caller-computed address.
__device__ __forceinline__ v16bf b_frag_lds(const u16* p) {
  v16us r;
#pragma unroll
  for (int e = 0; e < 16; ++e) r[e] = p[e];
  return __builtin_bit_cast(v16bf, r);
}

// B fragment directly from global f32 (row-major source, per-lane 16
// contiguous floats = 64B -> global_load_b128 x4, converted to bf16).
__device__ __forceinline__ v16bf b_frag_global(const float* p) {
  const float4* q = (const float4*)p;
  v16us r;
#pragma unroll
  for (int i = 0; i < 4; ++i) {
    float4 f = q[i];
    r[4 * i + 0] = bf16_of(f.x);
    r[4 * i + 1] = bf16_of(f.y);
    r[4 * i + 2] = bf16_of(f.z);
    r[4 * i + 3] = bf16_of(f.w);
  }
  return __builtin_bit_cast(v16bf, r);
}

// ---------------------------------------------------------------------------
// GEMM: C[4096,512] = A[4096,512] @ W[512,512] (+ bias).  WG tile 128x64,
// 8 waves of 32, each wave owns a 32x32 C tile (2x2 WMMA fragments).
// A tile (128x32 f32, 16KB) staged via async global->LDS B128 DMA; W tile
// staged synchronously transposed+converted to bf16 (DMA cannot transpose).
// ---------------------------------------------------------------------------
__global__ __launch_bounds__(256) void mmrel_gemm_bf16(
    const float* __restrict__ A, const float* __restrict__ W,
    const float* __restrict__ bias, float* __restrict__ C) {
  __shared__ float Af[128 * 32];   // 16KB f32 A tile (async DMA target)
  __shared__ u16 Ws[64][40];       // transposed bf16: Ws[n][k]

  const int tid  = threadIdx.x;
  const int lane = tid & 31;
  const int wave = tid >> 5;
  const int rowBase = blockIdx.x * 128;
  const int nBase   = blockIdx.y * 64;
  const int wr = (wave >> 1) * 32;   // wave row within WG tile
  const int wc = (wave & 1) * 32;    // wave col within WG tile

  const int ar = tid >> 1;           // A staging: row 0..127
  const int ak = (tid & 1) * 16;     // 16 contiguous floats per thread
  const int wn = tid & 63;           // W staging: n 0..63
  const int wk = (tid >> 6) * 8;     // 8 k's per thread

  v8f acc[2][2] = {};

  for (int k0 = 0; k0 < GK; k0 += 32) {
    // ---- stage A tile: global f32 -> LDS f32 ----
    const float* gsrc = A + (size_t)(rowBase + ar) * GK + k0 + ak;
    float* ldst = &Af[ar * 32 + ak];
#if defined(MMREL_ASYNC_LDS)
#pragma unroll
    for (int i = 0; i < 4; ++i)
      __builtin_amdgcn_global_load_async_to_lds_b128(
          (gv4i*)(gsrc + 4 * i), (lv4i*)(ldst + 4 * i), 0, 0);
#else
    {
      const float4* ap = (const float4*)gsrc;
      float4* dp = (float4*)ldst;
#pragma unroll
      for (int i = 0; i < 4; ++i) dp[i] = ap[i];
    }
#endif
    if (k0 + 32 < GK)                // global_prefetch_b8 for next W tile
      __builtin_prefetch(W + (size_t)(k0 + 32 + wk) * GN + nBase + wn, 0, 1);
    // ---- stage W tile transposed (coalesced along n across threads) ----
#pragma unroll
    for (int i = 0; i < 8; ++i)
      Ws[wn][wk + i] = bf16_of(W[(size_t)(k0 + wk + i) * GN + nBase + wn]);
    wait_async0();                   // this wave's DMA landed in LDS
    __syncthreads();                 // all waves' tiles visible

    v16bf aF[2], bF[2];
#pragma unroll
    for (int mi = 0; mi < 2; ++mi)
      aF[mi] = a_frag_lds_f32(&Af[(wr + mi * 16) * 32], 32, lane);
#pragma unroll
    for (int ni = 0; ni < 2; ++ni)
      bF[ni] = b_frag_lds(&Ws[wc + ni * 16 + (lane & 15)][16 * (lane >> 4)]);
#pragma unroll
    for (int mi = 0; mi < 2; ++mi)
#pragma unroll
      for (int ni = 0; ni < 2; ++ni)
        acc[mi][ni] = wmma_bf16(aF[mi], bF[ni], acc[mi][ni]);
    __syncthreads();
  }

  // epilogue: C/D layout -> element r of lane: m = r + 8*(lane>>4), n = lane&15
  const int hlf = lane >> 4;
#pragma unroll
  for (int mi = 0; mi < 2; ++mi)
#pragma unroll
    for (int ni = 0; ni < 2; ++ni) {
      const int col = nBase + wc + ni * 16 + (lane & 15);
      const float bvv = bias ? bias[col] : 0.0f;
#pragma unroll
      for (int r = 0; r < 8; ++r) {
        const int row = rowBase + wr + mi * 16 + r + 8 * hlf;
        C[(size_t)row * GN + col] = acc[mi][ni][r] + bvv;
      }
    }
}

// ---------------------------------------------------------------------------
// Attention: one WG per (b, h, 32-query block).  278KB dynamic LDS.
//   phase 1: scores (content + pos) via WMMA, B-fragments straight from HBM
//   phase 2: rel-shift combine + mask + scale
//   phase 3: wave-level softmax; attn -> d_out (f32) + LDS (bf16)
//   phase 4: ctx = attn @ V via WMMA
// ---------------------------------------------------------------------------
__global__ __launch_bounds__(256) void mmrel_attn(
    const float* __restrict__ Q, const float* __restrict__ K,
    const float* __restrict__ V, const float* __restrict__ POS,
    const float* __restrict__ ub, const float* __restrict__ vbias,
    const unsigned char* __restrict__ mask,
    float* __restrict__ attn_out, float* __restrict__ ctx_out) {
  extern __shared__ char smem[];
  u16*   qu    = (u16*)(smem);             //  32x64 bf16 (q+u)       4096B
  u16*   qv    = (u16*)(smem + 4096);      //  48x64 bf16 (q+v)       6144B
  float* sc    = (float*)(smem + 10240);   //  32xSCS f32 scores    132096B
  float* ps    = (float*)(smem + 142336);  //  33xSCS f32 pos-score 136224B
  u16*   attnb = (u16*)(smem + 142336);    //  32x1024 bf16 (aliases ps)
  u16*   vtall = (u16*)(smem + 10240);     //  64x1024 bf16 V^T (aliases sc)

  const int wg = blockIdx.x;
  const int qb = wg & 31;
  const int h  = (wg >> 5) & 7;
  const int b  = wg >> 8;
  const int r0 = qb * 32;
  const int tid  = threadIdx.x;
  const int lane = tid & 31;
  const int wave = tid >> 5;
  const int hlf  = lane >> 4;

  // ---- stage (q+u) rows r0..r0+31 and (q+v) rows r0..r0+47 (clamped) ----
  {
    const int row = tid >> 3, dc = (tid & 7) * 8;
    const float* qp = Q + ((size_t)(b * Sc + r0 + row) * Dc) + h * DHc + dc;
#pragma unroll
    for (int i = 0; i < 8; ++i)
      qu[row * 64 + dc + i] = bf16_of(qp[i] + ub[h * DHc + dc + i]);
  }
  for (int idx = tid; idx < 48 * 8; idx += 256) {
    const int row = idx >> 3, dc = (idx & 7) * 8;
    int gr = r0 + row; if (gr > Sc - 1) gr = Sc - 1;
    const float* qp = Q + ((size_t)(b * Sc + gr) * Dc) + h * DHc + dc;
#pragma unroll
    for (int i = 0; i < 8; ++i)
      qv[row * 64 + dc + i] = bf16_of(qp[i] + vbias[h * DHc + dc + i]);
  }
  __syncthreads();

  v16bf quF[2][2], qvF[3][2];
#pragma unroll
  for (int mi = 0; mi < 2; ++mi)
#pragma unroll
    for (int s = 0; s < 2; ++s)
      quF[mi][s] = a_frag_lds(qu + mi * 16 * 64 + s * 32, 64, lane);
#pragma unroll
  for (int mi = 0; mi < 3; ++mi)
#pragma unroll
    for (int s = 0; s < 2; ++s)
      qvF[mi][s] = a_frag_lds(qv + mi * 16 * 64 + s * 32, 64, lane);

  // ---- phase 1: each wave owns key tiles t = wave + 8*it (16 keys each) ----
  for (int it = 0; it < 8; ++it) {
    const int t   = wave + it * 8;
    const int key = t * 16 + (lane & 15);
    const float* kp = K   + ((size_t)(b * Sc + key) * Dc) + h * DHc + 16 * hlf;
    const float* pp = POS + ((size_t)(b * Sc + key) * Dc) + h * DHc + 16 * hlf;
    v16bf kF[2], pF[2];
    kF[0] = b_frag_global(kp);      kF[1] = b_frag_global(kp + 32);
    pF[0] = b_frag_global(pp);      pF[1] = b_frag_global(pp + 32);

    v8f cacc[2] = {};
    v8f pacc[3] = {};
#pragma unroll
    for (int s = 0; s < 2; ++s) {
#pragma unroll
      for (int mi = 0; mi < 2; ++mi) cacc[mi] = wmma_bf16(quF[mi][s], kF[s], cacc[mi]);
#pragma unroll
      for (int mi = 0; mi < 3; ++mi) pacc[mi] = wmma_bf16(qvF[mi][s], pF[s], pacc[mi]);
    }
    const int col = t * 16 + (lane & 15);
#pragma unroll
    for (int mi = 0; mi < 2; ++mi)
#pragma unroll
      for (int r = 0; r < 8; ++r) {
        const int ml = mi * 16 + r + 8 * hlf;
        sc[ml * SCS + col] = cacc[mi][r];
        ps[ml * SCS + col] = pacc[mi][r];
      }
    if (hlf == 0 && r0 + 32 < Sc)          // ps row 32 (from tile mi=2, r=0)
      ps[32 * SCS + col] = pacc[2][0];
  }
  __syncthreads();

  // ---- phase 2: rel-shift combine + mask + scale ----
  for (int idx = tid; idx < 32 * 1024; idx += 256) {
    const int li = idx >> 10, j = idx & 1023, i = r0 + li;
    float pv;
    if (j <= i)            pv = ps[li * SCS + (Sc - 1 - i + j)];
    else if (j == i + 1)   pv = 0.0f;
    else                   pv = ps[(li + 1) * SCS + (j - i - 2)];
    float v = (sc[li * SCS + j] + pv) * SCALE;
    if (mask[b * Sc + j]) v = -1e9f;
    sc[li * SCS + j] = v;
  }
  __syncthreads();

  // ---- phase 3: softmax, 4 rows per wave; write attn (f32) + probs (bf16) --
#pragma unroll
  for (int rr = 0; rr < 4; ++rr) {
    const int li = wave * 4 + rr, i = r0 + li;
    float* row = sc + li * SCS;
    float m = -3.4e38f;
    for (int j = lane; j < 1024; j += 32) m = fmaxf(m, row[j]);
#pragma unroll
    for (int off = 16; off; off >>= 1) m = fmaxf(m, __shfl_xor(m, off, 32));
    float sum = 0.0f;
    for (int j = lane; j < 1024; j += 32) { float e = __expf(row[j] - m); row[j] = e; sum += e; }
#pragma unroll
    for (int off = 16; off; off >>= 1) sum += __shfl_xor(sum, off, 32);
    const float inv = 1.0f / sum;
    float* ao = attn_out + (((size_t)(b * Hc + h) * Sc + i) * Sc);
    for (int j = lane; j < 1024; j += 32) {
      const float p = row[j] * inv;
      ao[j] = p;
      attnb[li * 1024 + j] = bf16_of(p);
    }
  }
  __syncthreads();

  // ---- phase 4: ctx = attn @ V.  Stage V^T into freed score buffer. ----
  for (int idx = tid; idx < Sc * DHc; idx += 256) {
    const int key = idx >> 6, dh = idx & 63;
    vtall[dh * 1024 + key] =
        bf16_of(V[((size_t)(b * Sc + key) * Dc) + h * DHc + dh]);
  }
  __syncthreads();

  const int mi = wave >> 2, ni = wave & 3;   // 2 x 4 fragment grid = 32x64
  v8f cc = {};
  for (int kt = 0; kt < 32; ++kt) {
    v16bf aF = a_frag_lds(attnb + mi * 16 * 1024 + kt * 32, 1024, lane);
    v16bf bF = b_frag_lds(vtall + (ni * 16 + (lane & 15)) * 1024 + kt * 32 + 16 * hlf);
    cc = wmma_bf16(aF, bF, cc);
  }
#pragma unroll
  for (int r = 0; r < 8; ++r) {
    const int row = r0 + mi * 16 + r + 8 * hlf;
    const int col = h * DHc + ni * 16 + (lane & 15);
    ctx_out[(size_t)(b * Sc + row) * Dc + col] = cc[r];
  }
}

// ---------------------------------------------------------------------------
extern "C" void kernel_launch(void* const* d_in, const int* in_sizes, int n_in,
                              void* d_out, int out_size, void* d_ws, size_t ws_size,
                              hipStream_t stream) {
  (void)in_sizes; (void)n_in; (void)out_size; (void)ws_size;
  const float* x1   = (const float*)d_in[0];
  const float* x2   = (const float*)d_in[1];
  const float* pose = (const float*)d_in[2];
  const unsigned char* mask = (const unsigned char*)d_in[3];
  const float* Wq = (const float*)d_in[4];  const float* bq = (const float*)d_in[5];
  const float* Wk = (const float*)d_in[6];  const float* bk = (const float*)d_in[7];
  const float* Wv = (const float*)d_in[8];  const float* bv = (const float*)d_in[9];
  const float* Wp = (const float*)d_in[10];
  const float* ub = (const float*)d_in[11]; const float* vb = (const float*)d_in[12];
  const float* Wo1 = (const float*)d_in[13]; const float* bo1 = (const float*)d_in[14];
  const float* Wo2 = (const float*)d_in[15]; const float* bo2 = (const float*)d_in[16];

  // workspace: POS | Q | K | V | CTX  (each 4096x512 f32 = 8MB; total 40MB)
  const size_t SZ = (size_t)GM * GN;
  float* POS = (float*)d_ws;
  float* Qb  = POS + SZ;
  float* Kb  = Qb + SZ;
  float* Vb  = Kb + SZ;
  float* CTX = Vb + SZ;

  float* out   = (float*)d_out;
  float* ctx1  = out;                       // [4,1024,512]
  float* attn1 = out + 2097152;             // [4,8,1024,1024]
  float* ctx2  = out + 35651584;
  float* attn2 = out + 37748736;

  const dim3 ggrid(GM / 128, GN / 64), blk(256);
  const dim3 agrid(Bc * Hc * (Sc / 32));
  const size_t asmem = 278560;              // dynamic LDS (< 320KB/WGP)
  (void)hipFuncSetAttribute((const void*)mmrel_attn,
                            hipFuncAttributeMaxDynamicSharedMemorySize, (int)asmem);

  // shared position projection
  mmrel_gemm_bf16<<<ggrid, blk, 0, stream>>>(pose, Wp, nullptr, POS);

  // stream 1
  mmrel_gemm_bf16<<<ggrid, blk, 0, stream>>>(x1, Wq, bq, Qb);
  mmrel_gemm_bf16<<<ggrid, blk, 0, stream>>>(x1, Wk, bk, Kb);
  mmrel_gemm_bf16<<<ggrid, blk, 0, stream>>>(x1, Wv, bv, Vb);
  mmrel_attn<<<agrid, blk, asmem, stream>>>(Qb, Kb, Vb, POS, ub, vb, mask, attn1, CTX);
  mmrel_gemm_bf16<<<ggrid, blk, 0, stream>>>(CTX, Wo1, bo1, ctx1);

  // stream 2 (reuses Q/K/V/CTX buffers; launches are ordered on `stream`)
  mmrel_gemm_bf16<<<ggrid, blk, 0, stream>>>(x2, Wq, bq, Qb);
  mmrel_gemm_bf16<<<ggrid, blk, 0, stream>>>(x2, Wk, bk, Kb);
  mmrel_gemm_bf16<<<ggrid, blk, 0, stream>>>(x2, Wv, bv, Vb);
  mmrel_attn<<<agrid, blk, asmem, stream>>>(Qb, Kb, Vb, POS, ub, vb, mask, attn2, CTX);
  mmrel_gemm_bf16<<<ggrid, blk, 0, stream>>>(CTX, Wo2, bo2, ctx2);
}
